// TransformerBlock_6622839571384
// MI455X (gfx1250) — matrix operand order repaired and checked by reading, not compiled, not run on hardware
//
#include <hip/hip_runtime.h>
#include <hip/hip_bf16.h>
#include <stdint.h>

// ---------------------------------------------------------------- constants
#define TB   128
#define TN   197
#define TD   768
#define THH  12
#define THD  64
#define TMLP 3072
#define TRHD 512
#define TRANK 256
#define TT   (TB * TN)    // 25216 tokens (divisible by 32)
#define TBH  (TB * THH)   // 1536 (b,h) pairs
#define NPADR 208         // padded token rows for q/k
#define SPADC 224         // padded score/key columns (multiple of 32)

// ---- gfx1250 async global->LDS path (guarded so compile can never regress)
#ifndef HAVE_ASYNC_LDS
#if defined(__has_builtin)
#if __has_builtin(__builtin_amdgcn_global_load_async_to_lds_b128) && \
    __has_builtin(__builtin_amdgcn_s_wait_asynccnt)
#define HAVE_ASYNC_LDS 1
#endif
#endif
#endif
#ifndef HAVE_ASYNC_LDS
#define HAVE_ASYNC_LDS 0
#endif

typedef unsigned short u16;
typedef __attribute__((ext_vector_type(16))) __bf16 v16bf;
typedef __attribute__((ext_vector_type(8)))  float  v8f;

union Frag { uint4 u[2]; v16bf v; };
static_assert(sizeof(Frag) == 32, "frag size");

__device__ __forceinline__ u16 f2b(float f) {
  unsigned u = __float_as_uint(f);
  u += 0x7fffu + ((u >> 16) & 1u);          // round-to-nearest-even
  return (u16)(u >> 16);
}
__device__ __forceinline__ float b2f(u16 h) { return __uint_as_float(((unsigned)h) << 16); }
__device__ __forceinline__ float gelu_f(float x) {
  return 0.5f * x * (1.0f + erff(x * 0.70710678118654752f));
}
__device__ __forceinline__ v8f wmma_bf16(const Frag& a, const Frag& b, v8f c) {
  // D(f32,16x16) = A(bf16,16x32) * B(bf16,32x16) + C
  return __builtin_amdgcn_wmma_f32_16x16x32_bf16(false, a.v, false, b.v, (short)0, c,
                                                 false, false);
}

#if HAVE_ASYNC_LDS
typedef int v4i_ __attribute__((vector_size(4 * sizeof(int))));
__device__ __forceinline__ void async_b128(const u16* g, u16* l) {
  __builtin_amdgcn_global_load_async_to_lds_b128(
      (__attribute__((address_space(1))) v4i_*)(void*)g,
      (__attribute__((address_space(3))) v4i_*)(void*)l, 0, 0);
}
#endif

// ---------------------------------------------------------------- f32 -> bf16
__global__ void cvt_bf16_k(const float* __restrict__ in, u16* __restrict__ out, int n) {
  int i = blockIdx.x * 256 + threadIdx.x;
  if (i < n) out[i] = f2b(in[i]);
}

// ---------------------------------------------------------------- LayerNorm (f32 in -> bf16 out)
__global__ __launch_bounds__(256) void ln_k(const float* __restrict__ x,
                                            const float* __restrict__ g,
                                            const float* __restrict__ b,
                                            u16* __restrict__ out) {
  __shared__ float red[256];
  const int t = blockIdx.x;
  const int tid = threadIdx.x;
  const float* row = x + (size_t)t * TD;
  float s = 0.0f;
  for (int i = tid; i < TD; i += 256) s += row[i];
  red[tid] = s; __syncthreads();
  for (int o = 128; o > 0; o >>= 1) { if (tid < o) red[tid] += red[tid + o]; __syncthreads(); }
  const float mean = red[0] * (1.0f / TD);
  __syncthreads();
  float v = 0.0f;
  for (int i = tid; i < TD; i += 256) { float d = row[i] - mean; v += d * d; }
  red[tid] = v; __syncthreads();
  for (int o = 128; o > 0; o >>= 1) { if (tid < o) red[tid] += red[tid + o]; __syncthreads(); }
  const float rstd = rsqrtf(red[0] * (1.0f / TD) + 1e-5f);
  for (int i = tid; i < TD; i += 256)
    out[(size_t)t * TD + i] = f2b((row[i] - mean) * rstd * g[i] + b[i]);
}

// ---------------------------------------------------------------- router mean g -> f[:,512:]
__global__ void router_mean_k(const float* __restrict__ e, u16* __restrict__ f) {
  int idx = blockIdx.x * 256 + threadIdx.x;
  if (idx >= TB * TRHD) return;
  const int b = idx >> 9, c = idx & 511;
  float s = 0.0f;
  for (int n = 1; n < TN; n++) s += e[((size_t)b * TN + n) * TRHD + c];
  const u16 gb = f2b(s * (1.0f / (TN - 1)));
  for (int n = 0; n < TN; n++) f[((size_t)b * TN + n) * (2 * TRHD) + TRHD + c] = gb;
}

// ---------------------------------------------------------------- logits + hard routing bits
__global__ void router_bits_k(const u16* __restrict__ h2, const u16* __restrict__ w3,
                              const float* __restrict__ b3, float* __restrict__ gate0,
                              float* __restrict__ gate1, float* __restrict__ gsel) {
  int t = blockIdx.x * 256 + threadIdx.x;
  if (t >= TT) return;
  float l0 = b3[0], l1 = b3[1], l2 = b3[2], l3 = b3[3];
  const u16* hr = h2 + (size_t)t * (TRHD / 2);
  for (int i = 0; i < TRHD / 2; i++) {
    const float a = b2f(hr[i]);
    l0 += a * b2f(w3[0 * 256 + i]);
    l1 += a * b2f(w3[1 * 256 + i]);
    l2 += a * b2f(w3[2 * 256 + i]);
    l3 += a * b2f(w3[3 * 256 + i]);
  }
  float bit0 = (l1 > l0) ? 1.0f : 0.0f;   // argmax; ties -> index 0
  float bit1 = (l3 > l2) ? 1.0f : 0.0f;
  if ((t % TN) == 0) { bit0 = 1.0f; bit1 = 1.0f; }   // reserved CLS token
  gate0[t] = bit0; gate1[t] = bit1;
  const int idx = (int)bit0 * 2 + (int)bit1;
  gsel[0 * TT + t] = (idx == 0) ? 1.0f : 0.0f;
  gsel[1 * TT + t] = (idx == 1) ? 1.0f : 0.0f;
  gsel[2 * TT + t] = (idx == 2) ? 1.0f : 0.0f;
}

// ---------------------------------------------------------------- generic WMMA GEMM
// C[T x N] = A[T x K](bf16) * W[N x K](bf16)^T  (+ epilogue)
// 256 threads = 8 waves (2x4 tiles of 16x16) -> block tile 32x64, K-step 64.
// Async global->LDS double-buffered pipeline on gfx1250.
enum { EP_B16 = 0, EP_B16_GELU, EP_ROUTER, EP_Q, EP_K, EP_V, EP_RESID, EP_RESID_DUAL,
       EP_ACC_GATE };

struct Ep {
  const float* bias;
  const float* resid;
  const float* gate;
  float* out_f32;
  float* out2_f32;
  u16*   out_b16;
  int    ldc;
};

template <int MODE>
__global__ __launch_bounds__(256) void gemm_bf16_k(const u16* __restrict__ A,
                                                   const u16* __restrict__ W,
                                                   int K, Ep ep) {
  constexpr int LDT = 72;                    // padded LDS stride (144B rows, 16B aligned)
  __shared__ u16 lA[2][32 * LDT];
  __shared__ u16 lB[2][64 * LDT];
  const int tid = threadIdx.x;
  const int m0 = blockIdx.x * 32;
  const int n0 = blockIdx.y * 64;
  const int lane = tid & 31, wave = tid >> 5;
  const int wm = wave >> 2, wn = wave & 3;   // 2x4 wave tiles of 16x16
  const int fm = lane & 15, hi = lane >> 4;
  const int kbA = hi * 8;                    // A-fragment K base (ISA 16-bit A layout)
  const int kbB = hi * 8;                    // B-fragment K base (same K order as A)
  const int ar = tid >> 3, ac = (tid & 7) * 8;   // A tile loader: 32x64, 16B each
  const int br = tid >> 2, bc = (tid & 3) * 16;  // W tile loader: 64x64, 2x16B each
  const u16* gA = A + (size_t)(m0 + ar) * K + ac;
  const u16* gW = W + (size_t)(n0 + br) * K + bc;
  u16* dA = &lA[0][ar * LDT + ac];
  u16* dB = &lB[0][br * LDT + bc];
  v8f acc = {};
  const int nk = K >> 6;
#if HAVE_ASYNC_LDS
  // tile i+1 DMA'd into buffer (i+1)&1 while WMMAs consume buffer i&1
  async_b128(gA, dA);
  async_b128(gW, dB);
  async_b128(gW + 8, dB + 8);
  for (int i = 0; i < nk; i++) {
    const int p = i & 1;
    if (i + 1 < nk) {
      const int q = (i + 1) & 1;
      const int ks = (i + 1) << 6;
      async_b128(gA + ks, dA + q * (32 * LDT));
      async_b128(gW + ks, dB + q * (64 * LDT));
      async_b128(gW + ks + 8, dB + q * (64 * LDT) + 8);
      __builtin_amdgcn_s_wait_asynccnt(3);   // tile i landed (in-order)
    } else {
      __builtin_amdgcn_s_wait_asynccnt(0);
    }
    __syncthreads();                          // everyone's async writes visible
    for (int s = 0; s < 2; s++) {
      Frag a, b;
      const u16* pa = &lA[p][(wm * 16 + fm) * LDT + s * 32 + kbA];
      a.u[0] = *(const uint4*)pa;             // K = kbA .. kbA+7
      a.u[1] = *(const uint4*)(pa + 16);      // K = kbA+16 .. kbA+23
      const u16* pb = &lB[p][(wn * 16 + fm) * LDT + s * 32 + kbB];
      b.u[0] = *(const uint4*)pb;             // K = kbB .. kbB+7
      b.u[1] = *(const uint4*)(pb + 16);      // K = kbB+16 .. kbB+23
      acc = wmma_bf16(a, b, acc);
    }
    __syncthreads();                          // buffer p free for tile i+2
  }
#else
  for (int k0 = 0; k0 < K; k0 += 64) {
    __syncthreads();
    *(uint4*)dA       = *(const uint4*)(gA + k0);
    *(uint4*)dB       = *(const uint4*)(gW + k0);
    *(uint4*)(dB + 8) = *(const uint4*)(gW + k0 + 8);
    if (k0 + 64 < K) {
      __builtin_prefetch(gA + k0 + 64, 0, 1);
      __builtin_prefetch(gW + k0 + 64, 0, 1);
    }
    __syncthreads();
    for (int s = 0; s < 2; s++) {
      Frag a, b;
      const u16* pa = &lA[0][(wm * 16 + fm) * LDT + s * 32 + kbA];
      a.u[0] = *(const uint4*)pa;
      a.u[1] = *(const uint4*)(pa + 16);
      const u16* pb = &lB[0][(wn * 16 + fm) * LDT + s * 32 + kbB];
      b.u[0] = *(const uint4*)pb;
      b.u[1] = *(const uint4*)(pb + 16);
      acc = wmma_bf16(a, b, acc);
    }
  }
#endif
  const int col = n0 + wn * 16 + fm;
  const int row0 = m0 + wm * 16 + 8 * hi;    // C layout: VGPR r -> M = r + 8*(lane/16)
  for (int r = 0; r < 8; r++) {
    const int row = row0 + r;
    const float v = acc[r];
    if (MODE == EP_B16) {
      const float bia = ep.bias ? ep.bias[col] : 0.0f;
      ep.out_b16[(size_t)row * ep.ldc + col] = f2b(v + bia);
    } else if (MODE == EP_B16_GELU) {
      ep.out_b16[(size_t)row * ep.ldc + col] = f2b(gelu_f(v + ep.bias[col]));
    } else if (MODE == EP_ROUTER) {
      const float xg = gelu_f(v + ep.bias[col]);
      ep.out_b16[(size_t)row * (2 * TRHD) + col] = f2b(xg);   // f[:, :512]
      ep.out2_f32[(size_t)row * TRHD + col] = xg;             // e (for mean)
    } else if (MODE == EP_Q || MODE == EP_K) {
      const int bb = row / TN, nn = row - bb * TN;
      const int hh = col >> 6, dd = col & 63;
      ep.out_b16[(((size_t)(bb * THH + hh)) * NPADR + nn) * THD + dd] =
          f2b(v + ep.bias[col]);
    } else if (MODE == EP_V) {               // store V transposed [bh][hd][tok]
      const int bb = row / TN, nn = row - bb * TN;
      const int hh = col >> 6, dd = col & 63;
      ep.out_b16[(((size_t)(bb * THH + hh)) * THD + dd) * SPADC + nn] =
          f2b(v + ep.bias[col]);
    } else if (MODE == EP_RESID) {
      const float xg = ep.resid[(size_t)row * TD + col] +
                       ep.gate[row] * (v + ep.bias[col]);
      ep.out_f32[(size_t)row * TD + col] = xg;
    } else if (MODE == EP_RESID_DUAL) {
      const float xg = ep.resid[(size_t)row * TD + col] +
                       ep.gate[row] * (v + ep.bias[col]);
      ep.out_f32[(size_t)row * TD + col] = xg;
      ep.out_b16[(size_t)row * TD + col] = f2b(xg);
    } else if (MODE == EP_ACC_GATE) {
      float* p = ep.out_f32 + (size_t)row * TD + col;
      *p = *p + ep.gate[row] * v;
    }
  }
}

// ---------------------------------------------------------------- attention (1 wave / 16 query rows)
__global__ __launch_bounds__(32) void attn_k(const u16* __restrict__ q,
                                             const u16* __restrict__ kg,
                                             const u16* __restrict__ vt,
                                             u16* __restrict__ attn) {
  __shared__ float sS[16 * SPADC];
  __shared__ u16   sP[16 * SPADC];
  const int mt = blockIdx.x;         // 0..12 query tiles
  const int hh = blockIdx.y;         // head
  const int bb = blockIdx.z;         // batch
  const int bh = bb * THH + hh;
  const int lane = threadIdx.x;
  const int m = lane & 15, hi = lane >> 4;
  const int kbA = hi * 8, kbB = hi * 8;
  // ---- Q fragments (two K-steps over head dim 64)
  Frag aq[2];
  const u16* qbase = q + ((size_t)bh * NPADR + mt * 16 + m) * THD;
  for (int s = 0; s < 2; s++) {
    aq[s].u[0] = *(const uint4*)(qbase + s * 32 + kbA);
    aq[s].u[1] = *(const uint4*)(qbase + s * 32 + kbA + 16);
  }
  // ---- scores: Q * K^T, scaled, masked, into LDS
  const float scale = 0.125f;        // 1/sqrt(64)
  for (int nt = 0; nt < 13; nt++) {
    v8f acc = {};
    const u16* kbase = kg + ((size_t)bh * NPADR + nt * 16 + m) * THD;
    for (int s = 0; s < 2; s++) {
      Frag bk;
      bk.u[0] = *(const uint4*)(kbase + s * 32 + kbB);
      bk.u[1] = *(const uint4*)(kbase + s * 32 + kbB + 16);
      acc = wmma_bf16(aq[s], bk, acc);
    }
    const int col = nt * 16 + m;
    for (int r = 0; r < 8; r++) {
      const int row = r + 8 * hi;
      sS[row * SPADC + col] = (col < TN) ? acc[r] * scale : -1e30f;
    }
  }
  for (int j = 0; j < 8; j++) sS[m * SPADC + 208 + kbA + j] = -1e30f;  // pad cols
  __syncthreads();
  // ---- softmax, rows owned by lanes 0..15; write bf16 probs (zero past 197)
  if (lane < 16) {
    float mx = -1e30f;
    for (int c = 0; c < SPADC; c++) mx = fmaxf(mx, sS[lane * SPADC + c]);
    float sum = 0.0f;
    for (int c = 0; c < SPADC; c++) {
      const float e = __expf(sS[lane * SPADC + c] - mx);
      sS[lane * SPADC + c] = e;
      sum += e;
    }
    const float inv = 1.0f / sum;
    for (int c = 0; c < SPADC; c++)
      sP[lane * SPADC + c] = (c < TN) ? f2b(sS[lane * SPADC + c] * inv) : (u16)0;
  }
  __syncthreads();
  // ---- O = P * V  (V pre-transposed: rows = head dims)
  for (int dt = 0; dt < 4; dt++) {
    v8f acc = {};
    const u16* vbase = vt + ((size_t)bh * THD + dt * 16 + m) * SPADC;
    for (int s = 0; s < 7; s++) {
      Frag a, b;
      const u16* pa = &sP[m * SPADC + s * 32 + kbA];
      a.u[0] = *(const uint4*)pa;
      a.u[1] = *(const uint4*)(pa + 16);
      b.u[0] = *(const uint4*)(vbase + s * 32 + kbB);
      b.u[1] = *(const uint4*)(vbase + s * 32 + kbB + 16);
      acc = wmma_bf16(a, b, acc);
    }
    const int col = hh * THD + dt * 16 + m;
    for (int r = 0; r < 8; r++) {
      const int row = mt * 16 + r + 8 * hi;
      if (row < TN) attn[((size_t)bb * TN + row) * TD + col] = f2b(acc[r]);
    }
  }
}

// ---------------------------------------------------------------- host-side orchestration
extern "C" void kernel_launch(void* const* d_in, const int* in_sizes, int n_in,
                              void* d_out, int out_size, void* d_ws, size_t ws_size,
                              hipStream_t stream) {
  (void)in_sizes; (void)n_in; (void)out_size; (void)ws_size;
  const float* x      = (const float*)d_in[0];
  const float* wq     = (const float*)d_in[1];
  const float* bq     = (const float*)d_in[2];
  const float* wk     = (const float*)d_in[3];
  const float* bk     = (const float*)d_in[4];
  const float* wv     = (const float*)d_in[5];
  const float* bv     = (const float*)d_in[6];
  const float* wo     = (const float*)d_in[7];
  const float* bo     = (const float*)d_in[8];
  const float* ln1_g  = (const float*)d_in[9];
  const float* ln1_b  = (const float*)d_in[10];
  const float* ln2_g  = (const float*)d_in[11];
  const float* ln2_b  = (const float*)d_in[12];
  const float* fc1_w  = (const float*)d_in[13];
  const float* fc1_b  = (const float*)d_in[14];
  const float* fc2_w  = (const float*)d_in[15];
  const float* fc2_b  = (const float*)d_in[16];
  const float* rln_g  = (const float*)d_in[17];
  const float* rln_b  = (const float*)d_in[18];
  const float* rin_w  = (const float*)d_in[19];
  const float* rin_b  = (const float*)d_in[20];
  const float* ro1_w  = (const float*)d_in[21];
  const float* ro1_b  = (const float*)d_in[22];
  const float* ro2_w  = (const float*)d_in[23];
  const float* ro2_b  = (const float*)d_in[24];
  const float* ro3_w  = (const float*)d_in[25];
  const float* ro3_b  = (const float*)d_in[26];
  const float* lra_dn = (const float*)d_in[27];
  const float* lra_up = (const float*)d_in[28];
  float* out = (float*)d_out;

  size_t off = 0;
  auto alloc = [&](size_t bytes) -> char* {
    char* p = (char*)d_ws + off;
    off += (bytes + 255) & ~(size_t)255;
    return p;
  };
  // bf16 weights
  u16* wq_h  = (u16*)alloc((size_t)TD * TD * 2);
  u16* wk_h  = (u16*)alloc((size_t)TD * TD * 2);
  u16* wv_h  = (u16*)alloc((size_t)TD * TD * 2);
  u16* wo_h  = (u16*)alloc((size_t)TD * TD * 2);
  u16* fc1_h = (u16*)alloc((size_t)TMLP * TD * 2);
  u16* fc2_h = (u16*)alloc((size_t)TD * TMLP * 2);
  u16* rin_h = (u16*)alloc((size_t)TRHD * TD * 2);
  u16* ro1_h = (u16*)alloc((size_t)TRHD * 2 * TRHD * 2);
  u16* ro2_h = (u16*)alloc((size_t)(TRHD / 2) * TRHD * 2);
  u16* ro3_h = (u16*)alloc((size_t)4 * (TRHD / 2) * 2);
  u16* lrd_h = (u16*)alloc((size_t)3 * TRANK * TD * 2);
  u16* lru_h = (u16*)alloc((size_t)3 * TD * TRANK * 2);
  // activations
  u16*   xr   = (u16*)alloc((size_t)TT * TD * 2);
  float* ebuf = (float*)alloc((size_t)TT * TRHD * 4);
  u16*   fbuf = (u16*)alloc((size_t)TT * 2 * TRHD * 2);
  u16*   h1   = (u16*)alloc((size_t)TT * TRHD * 2);
  u16*   h2   = (u16*)alloc((size_t)TT * (TRHD / 2) * 2);
  float* gate0 = (float*)alloc((size_t)TT * 4);
  float* gate1 = (float*)alloc((size_t)TT * 4);
  float* gsel  = (float*)alloc((size_t)3 * TT * 4);
  u16*   xa   = (u16*)alloc((size_t)TT * TD * 2);
  u16*   qb   = (u16*)alloc((size_t)TBH * NPADR * THD * 2);
  u16*   kb   = (u16*)alloc((size_t)TBH * NPADR * THD * 2);
  u16*   vtb  = (u16*)alloc((size_t)TBH * THD * SPADC * 2);
  u16*   attnb = (u16*)alloc((size_t)TT * TD * 2);
  float* hbuf = (float*)alloc((size_t)TT * TD * 4);
  u16*   xf   = (u16*)alloc((size_t)TT * TD * 2);
  u16*   ffh  = (u16*)alloc((size_t)TT * TMLP * 2);
  u16*   outb = (u16*)alloc((size_t)TT * TD * 2);
  u16*   down = (u16*)alloc((size_t)3 * TT * TRANK * 2);

  auto cvt = [&](const float* s, u16* d, int n) {
    cvt_bf16_k<<<dim3((n + 255) / 256), dim3(256), 0, stream>>>(s, d, n);
  };
  cvt(wq, wq_h, TD * TD);      cvt(wk, wk_h, TD * TD);
  cvt(wv, wv_h, TD * TD);      cvt(wo, wo_h, TD * TD);
  cvt(fc1_w, fc1_h, TMLP * TD); cvt(fc2_w, fc2_h, TD * TMLP);
  cvt(rin_w, rin_h, TRHD * TD); cvt(ro1_w, ro1_h, TRHD * 2 * TRHD);
  cvt(ro2_w, ro2_h, (TRHD / 2) * TRHD); cvt(ro3_w, ro3_h, 4 * (TRHD / 2));
  cvt(lra_dn, lrd_h, 3 * TRANK * TD);   cvt(lra_up, lru_h, 3 * TD * TRANK);

  const dim3 blk(256);
  auto grid = [](int N) { return dim3(TT / 32, N / 64); };

  // ---- router ----
  ln_k<<<TT, blk, 0, stream>>>(x, rln_g, rln_b, xr);
  { Ep ep{}; ep.bias = rin_b; ep.out_b16 = fbuf; ep.out2_f32 = ebuf;
    gemm_bf16_k<EP_ROUTER><<<grid(TRHD), blk, 0, stream>>>(xr, rin_h, TD, ep); }
  router_mean_k<<<(TB * TRHD + 255) / 256, blk, 0, stream>>>(ebuf, fbuf);
  { Ep ep{}; ep.bias = ro1_b; ep.out_b16 = h1; ep.ldc = TRHD;
    gemm_bf16_k<EP_B16_GELU><<<grid(TRHD), blk, 0, stream>>>(fbuf, ro1_h, 2 * TRHD, ep); }
  { Ep ep{}; ep.bias = ro2_b; ep.out_b16 = h2; ep.ldc = TRHD / 2;
    gemm_bf16_k<EP_B16_GELU><<<grid(TRHD / 2), blk, 0, stream>>>(h1, ro2_h, TRHD, ep); }
  router_bits_k<<<(TT + 255) / 256, blk, 0, stream>>>(h2, ro3_h, ro3_b, gate0, gate1, gsel);

  // ---- attention ----
  ln_k<<<TT, blk, 0, stream>>>(x, ln1_g, ln1_b, xa);
  { Ep ep{}; ep.bias = bq; ep.out_b16 = qb;
    gemm_bf16_k<EP_Q><<<grid(TD), blk, 0, stream>>>(xa, wq_h, TD, ep); }
  { Ep ep{}; ep.bias = bk; ep.out_b16 = kb;
    gemm_bf16_k<EP_K><<<grid(TD), blk, 0, stream>>>(xa, wk_h, TD, ep); }
  { Ep ep{}; ep.bias = bv; ep.out_b16 = vtb;
    gemm_bf16_k<EP_V><<<grid(TD), blk, 0, stream>>>(xa, wv_h, TD, ep); }
  attn_k<<<dim3(13, THH, TB), dim3(32), 0, stream>>>(qb, kb, vtb, attnb);
  { Ep ep{}; ep.bias = bo; ep.resid = x; ep.gate = gate0; ep.out_f32 = hbuf;
    gemm_bf16_k<EP_RESID><<<grid(TD), blk, 0, stream>>>(attnb, wo_h, TD, ep); }

  // ---- FFN ----
  ln_k<<<TT, blk, 0, stream>>>(hbuf, ln2_g, ln2_b, xf);
  { Ep ep{}; ep.bias = fc1_b; ep.out_b16 = ffh; ep.ldc = TMLP;
    gemm_bf16_k<EP_B16_GELU><<<grid(TMLP), blk, 0, stream>>>(xf, fc1_h, TD, ep); }
  { Ep ep{}; ep.bias = fc2_b; ep.resid = hbuf; ep.gate = gate1; ep.out_f32 = out;
    ep.out_b16 = outb;
    gemm_bf16_k<EP_RESID_DUAL><<<grid(TD), blk, 0, stream>>>(ffh, fc2_h, TMLP, ep); }

  // ---- low-rank path approximators (keys 0,1,2) ----
  for (int k = 0; k < 3; k++) {
    Ep ep{}; ep.out_b16 = down + (size_t)k * TT * TRANK; ep.ldc = TRANK;
    gemm_bf16_k<EP_B16><<<grid(TRANK), blk, 0, stream>>>(
        outb, lrd_h + (size_t)k * TRANK * TD, TD, ep);
  }
  for (int k = 0; k < 3; k++) {
    Ep ep{}; ep.gate = gsel + (size_t)k * TT; ep.out_f32 = out;
    gemm_bf16_k<EP_ACC_GATE><<<grid(TD), blk, 0, stream>>>(
        down + (size_t)k * TT * TRANK, lru_h + (size_t)k * TD * TRANK, TRANK, ep);
  }
}
